// FullyConnectGCLayer_53025666236858
// MI455X (gfx1250) — matrix operand order repaired and checked by reading, not compiled, not run on hardware
//
#include <hip/hip_runtime.h>

// Problem constants (fixed by the reference)
constexpr int kB = 512;   // batch
constexpr int kC = 1024;  // channels
constexpr int kN = 64;    // nodes
constexpr int kD = 128;   // diag channels

constexpr int kStrideX = 40;  // padded per-column k-stride (bf16) for x slabs in LDS
constexpr int kStrideA = 72;  // padded row stride (bf16) for staged A in LDS

typedef __attribute__((ext_vector_type(16))) __bf16 bf16x16;
typedef __attribute__((ext_vector_type(8)))  __bf16 bf16x8;
typedef __attribute__((ext_vector_type(8)))  float  f32x8;
typedef __attribute__((ext_vector_type(4)))  unsigned int u32x4;
typedef __attribute__((ext_vector_type(8)))  unsigned int u32x8;

// ---------------------------------------------------------------------------
// WMMA helper (CDNA5 v_wmma_f32_16x16x32_bf16, wave32)
// ---------------------------------------------------------------------------
static __device__ __forceinline__ f32x8 wmma_bf16(bf16x16 a, bf16x16 b, f32x8 c) {
  return __builtin_amdgcn_wmma_f32_16x16x32_bf16(false, a, false, b, (short)0, c,
                                                 false, false);
}

// A-matrix fragment (16x32 tile, 16-bit, row-major source, K contiguous):
// lane holds row M=lane&15; lanes 0-15 K = 0..7 & 16..23, lanes 16-31 K = 8..15 & 24..31.
static __device__ __forceinline__ bf16x16 load_a_frag(const __bf16* __restrict__ base,
                                                      int ldk, int lane) {
  int r  = lane & 15;
  int kb = (lane >> 4) * 8;
  const __bf16* p = base + (size_t)r * ldk + kb;
  union { bf16x16 v; struct { bf16x8 lo, hi; } s; } u;
  u.s.lo = *(const bf16x8*)(p);       // K = kb .. kb+7
  u.s.hi = *(const bf16x8*)(p + 16);  // K = kb+16 .. kb+23
  return u.v;
}

// B-matrix fragment (32x16 tile) from a K-CONTIGUOUS (column-major) LDS slab:
// xs[col * stride + k].  lanes 0-15 hold K=0..7 & 16..23 of col n, lanes 16-31 K=8..15 & 24..31.
static __device__ __forceinline__ bf16x16 load_b_frag_km(const __bf16* __restrict__ xs,
                                                         int stride, int lane, int ncol0) {
  int n  = ncol0 + (lane & 15);
  int kb = (lane >> 4) * 8;
  const __bf16* p = xs + (size_t)n * stride + kb;
  union { bf16x16 v; struct { bf16x8 lo, hi; } s; } u;
  u.s.lo = *(const bf16x8*)(p);      // K = kb .. kb+7
  u.s.hi = *(const bf16x8*)(p + 16);  // K = kb+16 .. kb+23
  return u.v;
}

// ---------------------------------------------------------------------------
// CDNA5 async / TDM staging helpers (inline asm; portable across toolchains)
// ---------------------------------------------------------------------------
static __device__ __forceinline__ void async_copy_b128(unsigned lds_byte_off,
                                                       const void* gsrc) {
  // LDS[lds_byte_off .. +15] = MEM[gsrc .. +15]   (per lane, ASYNCcnt tracked)
  asm volatile("global_load_async_to_lds_b128 %0, %1, off"
               :: "v"(lds_byte_off), "v"(gsrc) : "memory");
}
static __device__ __forceinline__ void wait_asynccnt0() {
  asm volatile("s_wait_asynccnt 0x0" ::: "memory");
}
static __device__ __forceinline__ void wait_asynccnt1() {
  asm volatile("s_wait_asynccnt 0x1" ::: "memory");
}

// TDM: load a 2D bf16 tile (tile_k x tile_n), K contiguous in memory with row
// stride stride_k elements, into LDS with 4 dwords of padding every 16 dwords
// (i.e. 32 bf16 of data -> 40 bf16 LDS stride).
static __device__ __forceinline__ void tdm_load_tile_pad40(unsigned lds_byte,
                                                           const void* gsrc,
                                                           unsigned tile_k, unsigned tile_n,
                                                           unsigned tensor_k, unsigned tensor_n,
                                                           unsigned stride_k) {
  unsigned long long ga = (unsigned long long)(uintptr_t)gsrc;
  u32x4 g0;
  g0[0] = 1u;                                   // count = 1 (valid descriptor)
  g0[1] = lds_byte;                             // lds_addr (bytes)
  g0[2] = (unsigned)(ga & 0xFFFFFFFFu);         // global_addr[31:0]
  g0[3] = (unsigned)((ga >> 32) & 0x1FFFFFFu)   // global_addr[56:32]
        | (2u << 30);                           // type = 2 ("image")
  u32x8 g1;
  g1[0] = (1u << 16)                            // data_size = 2 bytes
        | (1u << 20)                            // pad_enable
        | (3u << 22)                            // pad_interval: 16 dwords (64B)
        | (3u << 25);                           // pad_amount: 4 dwords (16B)
  g1[1] = (tensor_k & 0xFFFFu) << 16;           // tensor_dim0[15:0]
  g1[2] = (tensor_k >> 16) | ((tensor_n & 0xFFFFu) << 16);  // dim0[31:16] | dim1[15:0]
  g1[3] = (tensor_n >> 16) | (tile_k << 16);    // dim1[31:16] | tile_dim0
  g1[4] = tile_n;                               // tile_dim1 (tile_dim2 = 0)
  g1[5] = stride_k;                             // tensor_dim0_stride[31:0]
  g1[6] = 0u;                                   // stride hi / dim1_stride lo
  g1[7] = 0u;                                   // dim1_stride hi
  asm volatile("tensor_load_to_lds %0, %1" :: "s"(g0), "s"(g1) : "memory");
}

// ---------------------------------------------------------------------------
// K0a: generic f32 -> bf16 conversion (W, adj_w)
// ---------------------------------------------------------------------------
__global__ void k_convert_bf16(const float* __restrict__ in, __bf16* __restrict__ out,
                               int n) {
  int i = blockIdx.x * blockDim.x + threadIdx.x;
  if (i < n) out[i] = (__bf16)in[i];
}

// ---------------------------------------------------------------------------
// K0b: transpose-convert x to xt[b][n][c] (c contiguous, bf16) + row sums.
//      One block per batch; 16 chunks of 64 channels via LDS.
// ---------------------------------------------------------------------------
__global__ void k_xt(const float* __restrict__ x, __bf16* __restrict__ xt,
                     float* __restrict__ xsum) {
  __shared__ __align__(16) float xf[64 * kN];  // 16 KB: (c_local, n)
  int b = blockIdx.x, t = threadIdx.x;  // 256 threads
  for (int chunk = 0; chunk < kC / 64; ++chunk) {
    int c0 = chunk * 64;
    const float4* src = (const float4*)(x + ((size_t)b * kC + c0) * kN);
#pragma unroll
    for (int i = 0; i < 4; ++i) ((float4*)xf)[t + i * 256] = src[t + i * 256];
    __syncthreads();
    if (t < 64) {  // xsum over n for channel c0+t
      float s = 0.f;
      for (int n = 0; n < kN; ++n) s += xf[t * kN + n];
      xsum[(size_t)b * kC + c0 + t] = s;
    }
    // write 64 n-rows x 64 c (bf16), packed as uint4
    int n = t >> 2, q = t & 3;  // q selects 16 channels
#pragma unroll
    for (int half = 0; half < 2; ++half) {
      unsigned w[4];
#pragma unroll
      for (int jj = 0; jj < 4; ++jj) {
        float f0 = xf[(q * 16 + half * 8 + jj * 2 + 0) * kN + n];
        float f1 = xf[(q * 16 + half * 8 + jj * 2 + 1) * kN + n];
        unsigned short u0 = __builtin_bit_cast(unsigned short, (__bf16)f0);
        unsigned short u1 = __builtin_bit_cast(unsigned short, (__bf16)f1);
        w[jj] = (unsigned)u0 | ((unsigned)u1 << 16);
      }
      uint4 v = {w[0], w[1], w[2], w[3]};
      *(uint4*)(xt + (size_t)b * kN * kC + (size_t)n * kC + c0 + q * 16 + half * 8) = v;
    }
    __syncthreads();
  }
}

// ---------------------------------------------------------------------------
// K1: s[b,d] = sigmoid( dot(diag_w[d,:], xsum[b,:]) / 64 + diag_b[d] )
// ---------------------------------------------------------------------------
__global__ void k_sigv(const float* __restrict__ xsum, const float* __restrict__ diag_w,
                       const float* __restrict__ diag_b, float* __restrict__ sbuf) {
  __shared__ float xs[kC];
  int b = blockIdx.x, t = threadIdx.x;  // blockDim = 128
#pragma unroll
  for (int i = 0; i < kC / 128; ++i) xs[t + i * 128] = xsum[(long)b * kC + t + i * 128];
  __syncthreads();
  const float* wrow = diag_w + (long)t * kC;
  float acc = 0.f;
  for (int c = 0; c < kC; ++c) acc = fmaf(wrow[c], xs[c], acc);
  float v = acc * (1.0f / kN) + diag_b[t];
  sbuf[(long)b * kD + t] = 1.0f / (1.0f + __expf(-v));
}

// ---------------------------------------------------------------------------
// K2: xa[b] = adj_w @ x[b] + adj_b   (M=128, N=64, K=1024), one block/batch.
//     Double-buffered TDM staging (wave 0 issues; TENSORcnt in-order per wave).
// ---------------------------------------------------------------------------
__global__ void k_xa(const __bf16* __restrict__ adjw, const float* __restrict__ adjb,
                     const __bf16* __restrict__ xt, float* __restrict__ xa) {
  __shared__ __align__(16) __bf16 xs[2][kN * kStrideX];  // 2 x 5120 B padded slabs
  int b = blockIdx.x;
  int t = threadIdx.x, wave = t >> 5, lane = t & 31;
  const __bf16* xtb = xt + (size_t)b * kN * kC;
  unsigned ldsbuf[2] = {(unsigned)(uintptr_t)&xs[0][0], (unsigned)(uintptr_t)&xs[1][0]};
  f32x8 acc[4] = {};
  if (wave == 0)
    tdm_load_tile_pad40(ldsbuf[0], xtb, 32, kN, kC, kN, kC);  // slab 0
  for (int k0 = 0; k0 < kC; k0 += 32) {
    int cur = (k0 >> 5) & 1;
    if (wave == 0) {
      if (k0 + 32 < kC) {
        tdm_load_tile_pad40(ldsbuf[cur ^ 1], xtb + k0 + 32, 32, kN, kC, kN, kC);
        __builtin_amdgcn_s_wait_tensorcnt(1);  // slab k0 (older) has landed
      } else {
        __builtin_amdgcn_s_wait_tensorcnt(0);
      }
    }
    __syncthreads();
    bf16x16 a = load_a_frag(adjw + (size_t)(wave * 16) * kC + k0, kC, lane);
#pragma unroll
    for (int nt = 0; nt < 4; ++nt)
      acc[nt] = wmma_bf16(a, load_b_frag_km(xs[cur], kStrideX, lane, nt * 16), acc[nt]);
    __syncthreads();
  }
  float* outp = xa + (size_t)b * kD * kN;
#pragma unroll
  for (int nt = 0; nt < 4; ++nt)
#pragma unroll
    for (int j = 0; j < 8; ++j) {
      int r = wave * 16 + j + 8 * (lane >> 4);
      int c = nt * 16 + (lane & 15);
      outp[(size_t)r * kN + c] = acc[nt][j] + adjb[r];
    }
}

// ---------------------------------------------------------------------------
// K3: build + normalize A per batch; A is symmetric. Store bf16.
//     xa slab staged via async global->LDS copies.
// ---------------------------------------------------------------------------
__global__ void k_A(const float* __restrict__ xa, const float* __restrict__ sbuf,
                    __bf16* __restrict__ Abf) {
  __shared__ __align__(16) float xs[kD * kN];  // 32 KB
  __shared__ __align__(16) float As[kN * kN];  // 16 KB
  __shared__ float p[kN], g[kD], rs[kN];
  int b = blockIdx.x, t = threadIdx.x;  // 256 threads
  const float* xab = xa + (size_t)b * kD * kN;
  unsigned xsbase = (unsigned)(uintptr_t)&xs[0];
#pragma unroll
  for (int i = 0; i < 8; ++i) {  // 2048 chunks of 16B
    int id = t + i * 256;
    async_copy_b128(xsbase + id * 16, xab + id * 4);
  }
  if (t < kD) g[t] = sbuf[(size_t)b * kD + t] - 0.5f;
  wait_asynccnt0();
  __syncthreads();
  if (t < kN) {
    float acc = 0.f;
    for (int d = 0; d < kD; ++d) acc += xs[d * kN + t];
    p[t] = acc;
  }
  __syncthreads();
  for (int idx = t; idx < kN * kN; idx += 256) {
    int n = idx >> 6, m = idx & 63;
    float acc = 0.5f * p[n] * p[m];
    for (int d = 0; d < kD; ++d) acc = fmaf(g[d] * xs[d * kN + n], xs[d * kN + m], acc);
    acc = fmaxf(acc, 0.f);
    if (n == m) acc += 1.0f;
    As[idx] = acc;
  }
  __syncthreads();
  if (t < kN) {
    float acc = 0.f;
    for (int m = 0; m < kN; ++m) acc += As[t * kN + m];
    rs[t] = rsqrtf(acc);
  }
  __syncthreads();
  __bf16* Ab = Abf + (size_t)b * kN * kN;
  for (int idx = t; idx < kN * kN; idx += 256) {
    int n = idx >> 6, m = idx & 63;
    Ab[idx] = (__bf16)(As[idx] * rs[n] * rs[m]);
  }
}

// ---------------------------------------------------------------------------
// K4: Wx[b] = W @ x[b]   (M=1024 via 8 m-tiles, N=64, K=1024) -> bf16.
//     Double-buffered async global->LDS staging (ASYNCcnt, in-order per wave).
// ---------------------------------------------------------------------------
__global__ void k_wx(const __bf16* __restrict__ Wbf, const __bf16* __restrict__ xt,
                     __bf16* __restrict__ wx) {
  __shared__ __align__(16) __bf16 xs[2][kN * kStrideX];
  int b = blockIdx.y, mt = blockIdx.x;
  int t = threadIdx.x, wave = t >> 5, lane = t & 31;
  const __bf16* xtb = xt + (size_t)b * kN * kC;
  const __bf16* wrow = Wbf + (size_t)(mt * 128 + wave * 16) * kC;
  int n = t >> 2, q = t & 3;  // each thread stages one 16B chunk per slab
  unsigned ldsoff[2] = {(unsigned)(uintptr_t)&xs[0][n * kStrideX + q * 8],
                        (unsigned)(uintptr_t)&xs[1][n * kStrideX + q * 8]};
  const __bf16* src = xtb + (size_t)n * kC + q * 8;
  f32x8 acc[4] = {};
  async_copy_b128(ldsoff[0], src);  // slab 0
  for (int k0 = 0; k0 < kC; k0 += 32) {
    int cur = (k0 >> 5) & 1;
    if (k0 + 32 < kC) {
      async_copy_b128(ldsoff[cur ^ 1], src + k0 + 32);
      wait_asynccnt1();  // in-order: the older (slab k0) async has completed
    } else {
      wait_asynccnt0();
    }
    __syncthreads();
    bf16x16 a = load_a_frag(wrow + k0, kC, lane);
#pragma unroll
    for (int nt = 0; nt < 4; ++nt)
      acc[nt] = wmma_bf16(a, load_b_frag_km(xs[cur], kStrideX, lane, nt * 16), acc[nt]);
    __syncthreads();
  }
  __bf16* outp = wx + (size_t)b * kC * kN;
#pragma unroll
  for (int nt = 0; nt < 4; ++nt)
#pragma unroll
    for (int j = 0; j < 8; ++j) {
      int r = mt * 128 + wave * 16 + j + 8 * (lane >> 4);
      int c = nt * 16 + (lane & 15);
      outp[(size_t)r * kN + c] = (__bf16)acc[nt][j];
    }
}

// ---------------------------------------------------------------------------
// K5: out[b] = relu(Wx[b] @ A[b]) + x[b]  (M=1024 via 8 m-tiles, N=64, K=64).
//     A is symmetric -> B-fragments read K-contiguously; A staged via async.
// ---------------------------------------------------------------------------
__global__ void k_out(const __bf16* __restrict__ wx, const __bf16* __restrict__ Abf,
                      const float* __restrict__ x, float* __restrict__ out) {
  __shared__ __align__(16) __bf16 As[kN * kStrideA];  // padded 64x72 bf16
  int b = blockIdx.y, mt = blockIdx.x;
  int t = threadIdx.x, wave = t >> 5, lane = t & 31;
  const __bf16* abf = Abf + (size_t)b * kN * kN;
  unsigned asbase = (unsigned)(uintptr_t)&As[0];
#pragma unroll
  for (int i = 0; i < 2; ++i) {
    int id = t + i * 256;  // 512 chunks of 16B
    int m = id >> 3, c8 = id & 7;
    async_copy_b128(asbase + (m * kStrideA + c8 * 8) * 2, abf + id * 8);
  }
  wait_asynccnt0();
  __syncthreads();
  const __bf16* wrow = wx + (size_t)b * kC * kN + (size_t)(mt * 128 + wave * 16) * kN;
  f32x8 acc[4] = {};
#pragma unroll
  for (int k0 = 0; k0 < kN; k0 += 32) {
    bf16x16 a = load_a_frag(wrow + k0, kN, lane);
#pragma unroll
    for (int nt = 0; nt < 4; ++nt)
      acc[nt] = wmma_bf16(a, load_b_frag_km(As + k0, kStrideA, lane, nt * 16), acc[nt]);
  }
#pragma unroll
  for (int nt = 0; nt < 4; ++nt)
#pragma unroll
    for (int j = 0; j < 8; ++j) {
      int r = mt * 128 + wave * 16 + j + 8 * (lane >> 4);
      int c = nt * 16 + (lane & 15);
      size_t o = (size_t)b * kC * kN + (size_t)r * kN + c;
      out[o] = fmaxf(acc[nt][j], 0.f) + x[o];
    }
}

// ---------------------------------------------------------------------------
extern "C" void kernel_launch(void* const* d_in, const int* in_sizes, int n_in,
                              void* d_out, int out_size, void* d_ws, size_t ws_size,
                              hipStream_t stream) {
  (void)in_sizes; (void)n_in; (void)out_size; (void)ws_size;
  const float* x      = (const float*)d_in[0];
  const float* adj_w  = (const float*)d_in[1];
  const float* adj_b  = (const float*)d_in[2];
  const float* diag_w = (const float*)d_in[3];
  const float* diag_b = (const float*)d_in[4];
  const float* W      = (const float*)d_in[5];
  float* out = (float*)d_out;

  char* ws = (char*)d_ws;
  size_t off = 0;
  auto take = [&](size_t bytes) -> char* {
    char* p = ws + off;
    off = (off + bytes + 255) & ~(size_t)255;
    return p;
  };
  __bf16* xt   = (__bf16*)take((size_t)kB * kN * kC * 2);  // 67 MB  (b, n, c)
  __bf16* Wbf  = (__bf16*)take((size_t)kC * kC * 2);       // 2 MB
  __bf16* awbf = (__bf16*)take((size_t)kD * kC * 2);       // 256 KB
  float*  xsum = (float*) take((size_t)kB * kC * 4);       // 2 MB
  float*  sbuf = (float*) take((size_t)kB * kD * 4);       // 256 KB
  float*  xa   = (float*) take((size_t)kB * kD * kN * 4);  // 16.8 MB
  __bf16* Abf  = (__bf16*)take((size_t)kB * kN * kN * 2);  // 4.2 MB
  __bf16* wx   = (__bf16*)take((size_t)kB * kC * kN * 2);  // 67 MB

  k_convert_bf16<<<(kC * kC + 255) / 256, 256, 0, stream>>>(W, Wbf, kC * kC);
  k_convert_bf16<<<(kD * kC + 255) / 256, 256, 0, stream>>>(adj_w, awbf, kD * kC);
  k_xt<<<kB, 256, 0, stream>>>(x, xt, xsum);
  k_sigv<<<kB, 128, 0, stream>>>(xsum, diag_w, diag_b, sbuf);
  k_xa<<<kB, 256, 0, stream>>>(awbf, adj_b, xt, xa);
  k_A<<<kB, 256, 0, stream>>>(xa, sbuf, Abf);
  k_wx<<<dim3(8, kB), 256, 0, stream>>>(Wbf, xt, wx);
  k_out<<<dim3(8, kB), 256, 0, stream>>>(wx, Abf, x, out);
}
